// BatchContrastiveLoss_33320356282454
// MI455X (gfx1250) — compile-verified, matrix operand order checked by reading, not run on hardware
//
#include <hip/hip_runtime.h>
#include <math.h>

#define BS 4096
#define D  768
#define NCHUNKS 4
#define CHUNK_N (BS / NCHUNKS)    // 1024 columns per workgroup
#define WAVES 8
#define MT 2                      // M-tiles (16 rows each) per workgroup
#define NT 4                      // N-tiles per wave per inner pass
#define ROWS_PER_WG (16 * MT)     // 32
#define GROUPS (CHUNK_N / 16 / (WAVES * NT))   // 2 passes of 4 tiles per wave

typedef __attribute__((ext_vector_type(16))) __bf16 v16bf;
typedef __attribute__((ext_vector_type(8)))  __bf16 v8bf;
typedef __attribute__((ext_vector_type(4)))  __bf16 v4bf;
typedef __attribute__((ext_vector_type(8)))  float  v8f;

union Frag { v16bf v; v8bf h[2]; };

// ---------------------------------------------------------------------------
// Kernel 1: fp32 -> bf16 conversion of both matrices (memory-bound, ~2 us)
// ---------------------------------------------------------------------------
__global__ __launch_bounds__(256) void cvt_bf16(
    const float* __restrict__ code_f, const float* __restrict__ nl_f,
    __bf16* __restrict__ code_b, __bf16* __restrict__ nl_b)
{
    const int n = BS * D;
    int i = (blockIdx.x * blockDim.x + threadIdx.x) * 4;
    if (i < n) {
        float4 fc = *(const float4*)(code_f + i);
        float4 fn = *(const float4*)(nl_f + i);
        v4bf oc, on;
        oc[0] = (__bf16)fc.x; oc[1] = (__bf16)fc.y;
        oc[2] = (__bf16)fc.z; oc[3] = (__bf16)fc.w;
        on[0] = (__bf16)fn.x; on[1] = (__bf16)fn.y;
        on[2] = (__bf16)fn.z; on[3] = (__bf16)fn.w;
        *(v4bf*)(code_b + i) = oc;
        *(v4bf*)(nl_b + i)   = on;
    }
}

// Store the diagonal element of a 16x16 C tile whose N-base equals its M-base.
// C layout: lane holds (M = j + 8*(lane>>4), N = lane&15) in acc[j].
__device__ __forceinline__ void store_diag(const v8f& a, int rowBase, int lane,
                                           float* __restrict__ diag)
{
    const int half = lane >> 4;
#pragma unroll
    for (int j = 0; j < 8; ++j) {
        const int M = j + (half << 3);
        if ((lane & 15) == M) diag[rowBase + M] = a[j];
    }
}

// ---------------------------------------------------------------------------
// Kernel 2: WMMA GEMM (2x4 register tile per wave) + fused online logsumexp
//   grid = (BS/32, NCHUNKS), block = 256 (8 waves, wave32)
//   A = nl (rows of scores), B = code (scores[n][m] = nl[n] . code[m])
// ---------------------------------------------------------------------------
__global__ __launch_bounds__(256) void gemm_lse(
    const __bf16* __restrict__ A, const __bf16* __restrict__ B,
    float* __restrict__ pm, float* __restrict__ ps, float* __restrict__ diag)
{
    const int lane   = threadIdx.x & 31;
    const int wave   = threadIdx.x >> 5;
    const int mbase  = blockIdx.x * ROWS_PER_WG;
    const int nchunk = blockIdx.y;
    const int nbase0 = nchunk * CHUNK_N;

    // 16-bit A-matrix layout: lane%16 = row M, lane/16 selects K+8 subchunk,
    // VGPR0-3 hold K..K+7(+8), VGPR4-7 hold K+16..K+23(+8).
    const int l15  = lane & 15;
    const int koff = (lane >> 4) << 3;                 // 0 or 8
    const __bf16* aRow0 = A + (size_t)(mbase + l15) * D + koff;
    const __bf16* aRow1 = aRow0 + (size_t)16 * D;

    // m/s index r = mt*8 + j  <->  local row mt*16 + (lane>>4)*8 + j
    float m[MT * 8], s[MT * 8];
#pragma unroll
    for (int r = 0; r < MT * 8; ++r) { m[r] = -__builtin_inff(); s[r] = 0.f; }

    for (int g = 0; g < GROUPS; ++g) {
        const int gbase = nbase0 + (wave * (GROUPS * NT) + g * NT) * 16;
        const __bf16* bRow[NT];
#pragma unroll
        for (int c = 0; c < NT; ++c)
            bRow[c] = B + (size_t)(gbase + c * 16 + l15) * D + koff;

        v8f acc[MT * NT];
#pragma unroll
        for (int i = 0; i < MT * NT; ++i) acc[i] = (v8f){};

#pragma unroll 2
        for (int k = 0; k < D; k += 32) {
            Frag a0, a1, b[NT];
            a0.h[0] = *(const v8bf*)(aRow0 + k);
            a0.h[1] = *(const v8bf*)(aRow0 + k + 16);
            a1.h[0] = *(const v8bf*)(aRow1 + k);
            a1.h[1] = *(const v8bf*)(aRow1 + k + 16);
#pragma unroll
            for (int c = 0; c < NT; ++c) {
                b[c].h[0] = *(const v8bf*)(bRow[c] + k);
                b[c].h[1] = *(const v8bf*)(bRow[c] + k + 16);
            }
#pragma unroll
            for (int c = 0; c < NT; ++c)
                acc[c] = __builtin_amdgcn_wmma_f32_16x16x32_bf16(
                             false, a0.v, false, b[c].v, (short)0, acc[c],
                             false, false);
#pragma unroll
            for (int c = 0; c < NT; ++c)
                acc[NT + c] = __builtin_amdgcn_wmma_f32_16x16x32_bf16(
                                  false, a1.v, false, b[c].v, (short)0,
                                  acc[NT + c], false, false);
        }

        // Grouped logsumexp fold over the 4 fresh column samples per row
#pragma unroll
        for (int mt = 0; mt < MT; ++mt) {
#pragma unroll
            for (int j = 0; j < 8; ++j) {
                const int r = mt * 8 + j;
                float v0 = acc[mt * NT + 0][j], v1 = acc[mt * NT + 1][j];
                float v2 = acc[mt * NT + 2][j], v3 = acc[mt * NT + 3][j];
                float mx = fmaxf(fmaxf(v0, v1), fmaxf(v2, v3));
                float nm = fmaxf(m[r], mx);
                s[r] = s[r] * __expf(m[r] - nm)
                     + __expf(v0 - nm) + __expf(v1 - nm)
                     + __expf(v2 - nm) + __expf(v3 - nm);
                m[r] = nm;
            }
        }

        // Diagonal tiles (at most one per M-tile per workgroup)
#pragma unroll
        for (int c = 0; c < NT; ++c) {
            const int nt = gbase + c * 16;
            if (nt == mbase)      store_diag(acc[c],      mbase,      lane, diag);
            if (nt == mbase + 16) store_diag(acc[NT + c], mbase + 16, lane, diag);
        }
    }

    // Reduce across the 16 columns (lanes) of each half-wave
#pragma unroll
    for (int off = 1; off < 16; off <<= 1) {
#pragma unroll
        for (int r = 0; r < MT * 8; ++r) {
            float om = __shfl_xor(m[r], off, 32);
            float os = __shfl_xor(s[r], off, 32);
            float nm = fmaxf(m[r], om);
            s[r] = s[r] * __expf(m[r] - nm) + os * __expf(om - nm);
            m[r] = nm;
        }
    }

    // Combine the 8 waves (same 32 rows, disjoint columns) via LDS
    __shared__ float sm[WAVES][ROWS_PER_WG];
    __shared__ float ss[WAVES][ROWS_PER_WG];
    if ((lane & 15) == 0) {
        const int half = lane >> 4;
#pragma unroll
        for (int mt = 0; mt < MT; ++mt) {
#pragma unroll
            for (int j = 0; j < 8; ++j) {
                const int row = mt * 16 + (half << 3) + j;
                sm[wave][row] = m[mt * 8 + j];
                ss[wave][row] = s[mt * 8 + j];
            }
        }
    }
    __syncthreads();

    if (threadIdx.x < ROWS_PER_WG) {
        const int r = threadIdx.x;
        float M = -__builtin_inff(), S = 0.f;
#pragma unroll
        for (int w = 0; w < WAVES; ++w) {
            float om = sm[w][r], os = ss[w][r];
            float nm = fmaxf(M, om);
            S = S * __expf(M - nm) + os * __expf(om - nm);
            M = nm;
        }
        pm[(size_t)(mbase + r) * NCHUNKS + nchunk] = M;
        ps[(size_t)(mbase + r) * NCHUNKS + nchunk] = S;
    }
}

// ---------------------------------------------------------------------------
// Kernel 3: merge chunk partials per row, subtract diagonal, mean -> scalar
// ---------------------------------------------------------------------------
__global__ __launch_bounds__(256) void finalize_loss(
    const float* __restrict__ pm, const float* __restrict__ ps,
    const float* __restrict__ diag, float* __restrict__ out)
{
    __shared__ float red[256];
    float local = 0.f;
    for (int r = threadIdx.x; r < BS; r += 256) {
        float M = -__builtin_inff(), S = 0.f;
#pragma unroll
        for (int c = 0; c < NCHUNKS; ++c) {
            float om = pm[r * NCHUNKS + c], os = ps[r * NCHUNKS + c];
            float nm = fmaxf(M, om);
            S = S * __expf(M - nm) + os * __expf(om - nm);
            M = nm;
        }
        local += (M + __logf(S)) - diag[r];
    }
    red[threadIdx.x] = local;
    __syncthreads();
    for (int st = 128; st > 0; st >>= 1) {
        if (threadIdx.x < st) red[threadIdx.x] += red[threadIdx.x + st];
        __syncthreads();
    }
    if (threadIdx.x == 0) out[0] = red[0] / (float)BS;
}

// ---------------------------------------------------------------------------
extern "C" void kernel_launch(void* const* d_in, const int* in_sizes, int n_in,
                              void* d_out, int out_size, void* d_ws, size_t ws_size,
                              hipStream_t stream)
{
    (void)in_sizes; (void)n_in; (void)out_size; (void)ws_size;
    const float* code_f = (const float*)d_in[0];   // code_vec [BS, D]
    const float* nl_f   = (const float*)d_in[1];   // nl_vec   [BS, D]
    float* out = (float*)d_out;

    // Workspace carve-up (all 256B-aligned regions)
    const size_t MAT_BYTES = (size_t)BS * D * sizeof(__bf16);   // 6,291,456
    char* ws = (char*)d_ws;
    __bf16* code_b = (__bf16*)(ws);
    __bf16* nl_b   = (__bf16*)(ws + MAT_BYTES);
    float*  pm     = (float*)(ws + 2 * MAT_BYTES);
    float*  ps     = pm + (size_t)BS * NCHUNKS;
    float*  diag   = ps + (size_t)BS * NCHUNKS;

    // 1) fp32 -> bf16
    {
        const int n = BS * D;                 // 3,145,728
        const int threads = n / 4;            // 786,432
        cvt_bf16<<<threads / 256, 256, 0, stream>>>(code_f, nl_f, code_b, nl_b);
    }
    // 2) WMMA GEMM with 2x4 register tiling + fused online logsumexp
    {
        dim3 grid(BS / ROWS_PER_WG, NCHUNKS); // (128, 4)
        gemm_lse<<<grid, 256, 0, stream>>>(nl_b, code_b, pm, ps, diag);
    }
    // 3) final scalar
    finalize_loss<<<1, 256, 0, stream>>>(pm, ps, diag, out);
}
